// CausalGatedD2Attention_18502719111278
// MI455X (gfx1250) — matrix operand order repaired and checked by reading, not compiled, not run on hardware
//
#include <hip/hip_runtime.h>

// ---------------------------------------------------------------------------
// CausalGatedD2Attention for MI455X (gfx1250, wave32, WMMA).
// All contractions -> v_wmma_f32_16x16x32_f16 (f16 in, f32 acc).
// Linear attention cumsum recurrence -> chunked form (chunk=64) so the
// sequential scan becomes per-chunk WMMA GEMMs.
// GEMM: 128x128 block tile, 8 waves, 32x64 per wave = 8 WMMA per K-step.
// ---------------------------------------------------------------------------

typedef __attribute__((ext_vector_type(16))) _Float16 v16h;
typedef __attribute__((ext_vector_type(8)))  _Float16 v8h;
typedef __attribute__((ext_vector_type(8)))  float    v8f;

#define LSEQ 2048
#define DMODEL 768
#define NHEAD 12
#define DHEAD 64

__device__ __forceinline__ v8f wmma_f16(v16h a, v16h b, v8f c) {
  // (neg_a, A, neg_b, B, c_mod, C, reuse_a, reuse_b)
  return __builtin_amdgcn_wmma_f32_16x16x32_f16(false, a, false, b, (short)0, c,
                                                false, false);
}

// A-operand fragment (16x32 f16, M=lane%16) from a row-major [rows][ld] LDS
// tile.  Per ISA layout: half h holds K = (h%8) + 8*(lane/16) + 16*(h/8)
// -> two contiguous 8-half (16B) runs per lane.
__device__ __forceinline__ v16h frag_a(const _Float16* base, int ld, int m,
                                       int k0, int lane) {
  const _Float16* p = base + m * ld + k0 + 8 * (lane >> 4);
  v8h lo = *(const v8h*)p;
  v8h hi = *(const v8h*)(p + 16);
  v16h r;
#pragma unroll
  for (int i = 0; i < 8; ++i) { r[i] = lo[i]; r[i + 8] = hi[i]; }
  return r;
}

// B-operand fragment (32x16 f16, N=lane%16) gathered from an LDS tile stored
// TRANSPOSED as [N][ld] so half h holds K = (h%8) + 8*(lane/16) + 16*(h/8):
// two contiguous 8-half (16B) runs per lane.
__device__ __forceinline__ v16h frag_b(const _Float16* baseT, int ld, int n,
                                       int k0, int lane) {
  const _Float16* p = baseT + n * ld + k0 + 8 * (lane >> 4);
  v8h lo = *(const v8h*)p;
  v8h hi = *(const v8h*)(p + 16);
  v16h r;
#pragma unroll
  for (int i = 0; i < 8; ++i) { r[i] = lo[i]; r[i + 8] = hi[i]; }
  return r;
}

// ---------------------------------------------------------------------------
// Kernel: f32 -> f16 conversion (weights).
// ---------------------------------------------------------------------------
__global__ __launch_bounds__(256) void cvt_f16_kernel(const float* __restrict__ s,
                                                      _Float16* __restrict__ d,
                                                      int n) {
  int i = blockIdx.x * 256 + threadIdx.x;
  if (i < n) d[i] = (_Float16)s[i];
}

// ---------------------------------------------------------------------------
// Kernel: LayerNorm of x; emits x (f16) and x_norm (f16).  One block per row.
// ---------------------------------------------------------------------------
__global__ __launch_bounds__(256) void layernorm_cvt_kernel(
    const float* __restrict__ x, const float* __restrict__ g,
    const float* __restrict__ bv, _Float16* __restrict__ xh,
    _Float16* __restrict__ xnh) {
  __shared__ float red[256];
  const int row = blockIdx.x;
  const int t = threadIdx.x;
  const float* xr = x + (size_t)row * DMODEL;
  float l0 = xr[t], l1 = xr[t + 256], l2 = xr[t + 512];
  red[t] = l0 + l1 + l2;
  __syncthreads();
  for (int st = 128; st > 0; st >>= 1) {
    if (t < st) red[t] += red[t + st];
    __syncthreads();
  }
  float mu = red[0] * (1.0f / DMODEL);
  __syncthreads();
  float d0 = l0 - mu, d1 = l1 - mu, d2 = l2 - mu;
  red[t] = d0 * d0 + d1 * d1 + d2 * d2;
  __syncthreads();
  for (int st = 128; st > 0; st >>= 1) {
    if (t < st) red[t] += red[t + st];
    __syncthreads();
  }
  float inv = rsqrtf(red[0] * (1.0f / DMODEL) + 1e-5f);
  size_t base = (size_t)row * DMODEL;
  xh[base + t]        = (_Float16)l0;
  xh[base + t + 256]  = (_Float16)l1;
  xh[base + t + 512]  = (_Float16)l2;
  xnh[base + t]       = (_Float16)(d0 * inv * g[t] + bv[t]);
  xnh[base + t + 256] = (_Float16)(d1 * inv * g[t + 256] + bv[t + 256]);
  xnh[base + t + 512] = (_Float16)(d2 * inv * g[t + 512] + bv[t + 512]);
}

// ---------------------------------------------------------------------------
// Kernel: WMMA GEMM  C[MxN] = A_f16[MxK] @ B_f16[KxN] + bias.
// Block tile 128x128, 8 waves as 4(M) x 2(N); each wave 32x64 -> 8 WMMA/K-step.
// ---------------------------------------------------------------------------
__global__ __launch_bounds__(256) void gemm_wmma_kernel(
    const _Float16* __restrict__ A, const _Float16* __restrict__ B,
    const float* __restrict__ bias, float* __restrict__ C, int M, int N,
    int K) {
  __shared__ __align__(32) _Float16 sA[128][32];   // [M][K]
  __shared__ __align__(32) _Float16 sBT[128][32];  // [N][K] (transposed)
  const int tid = threadIdx.x;
  const int wave = tid >> 5, lane = tid & 31;
  const int bm = blockIdx.x * 128, bn = blockIdx.y * 128;
  const int wm = (wave & 3) * 32;   // wave's 32-row M slab
  const int wn = (wave >> 2) * 64;  // wave's 64-col N slab
  v8f acc[2][4] = {};

  const int ar = tid >> 1, ac = (tid & 1) * 16;   // A tile load map (16 halves)
  const int bk = tid >> 3, bnn = (tid & 7) * 16;  // B tile load map (16 halves)

  for (int k0 = 0; k0 < K; k0 += 32) {
    // stage A tile (row-major) and B tile (transposed) in LDS
    {
      const _Float16* ap = &A[(size_t)(bm + ar) * K + k0 + ac];
      *(v8h*)&sA[ar][ac]     = *(const v8h*)ap;
      *(v8h*)&sA[ar][ac + 8] = *(const v8h*)(ap + 8);
      const _Float16* bp = &B[(size_t)(k0 + bk) * N + bn + bnn];
      v8h t0 = *(const v8h*)bp;
      v8h t1 = *(const v8h*)(bp + 8);
#pragma unroll
      for (int j = 0; j < 8; ++j) {
        sBT[bnn + j][bk]     = t0[j];
        sBT[bnn + 8 + j][bk] = t1[j];
      }
      // branchless prefetch of next K tile (clamped) -> global_prefetch_b8
      int kp = (k0 + 32 < K) ? (k0 + 32) : k0;
      __builtin_prefetch(&A[(size_t)(bm + ar) * K + kp + ac], 0, 0);
      __builtin_prefetch(&B[(size_t)(kp + bk) * N + bn + bnn], 0, 0);
    }
    __syncthreads();
    v16h a0 = frag_a(&sA[0][0], 32, wm + (lane & 15), 0, lane);
    v16h a1 = frag_a(&sA[0][0], 32, wm + 16 + (lane & 15), 0, lane);
    v16h b0 = frag_b(&sBT[0][0], 32, wn + (lane & 15), 0, lane);
    v16h b1 = frag_b(&sBT[0][0], 32, wn + 16 + (lane & 15), 0, lane);
    v16h b2 = frag_b(&sBT[0][0], 32, wn + 32 + (lane & 15), 0, lane);
    v16h b3 = frag_b(&sBT[0][0], 32, wn + 48 + (lane & 15), 0, lane);
    acc[0][0] = wmma_f16(a0, b0, acc[0][0]);
    acc[0][1] = wmma_f16(a0, b1, acc[0][1]);
    acc[0][2] = wmma_f16(a0, b2, acc[0][2]);
    acc[0][3] = wmma_f16(a0, b3, acc[0][3]);
    acc[1][0] = wmma_f16(a1, b0, acc[1][0]);
    acc[1][1] = wmma_f16(a1, b1, acc[1][1]);
    acc[1][2] = wmma_f16(a1, b2, acc[1][2]);
    acc[1][3] = wmma_f16(a1, b3, acc[1][3]);
    __syncthreads();
  }
#pragma unroll
  for (int i = 0; i < 2; ++i) {
    const int mb = bm + wm + i * 16 + 8 * (lane >> 4);
#pragma unroll
    for (int j = 0; j < 4; ++j) {
      const int n = bn + wn + j * 16 + (lane & 15);
      const float bi = bias[n];
#pragma unroll
      for (int r = 0; r < 8; ++r)
        C[(size_t)(mb + r) * N + n] = acc[i][j][r] + bi;
    }
  }
}

// ---------------------------------------------------------------------------
// Kernel: gate + feature maps.  Emits q_f,k_f,v in [H][L][DH] f16 layout.
// ---------------------------------------------------------------------------
__device__ __forceinline__ float softplusf(float v) {
  return v > 20.0f ? v : log1pf(expf(v));
}

__global__ __launch_bounds__(256) void gate_feat_kernel(
    const float* __restrict__ qkv, const float* __restrict__ sem,
    const float* __restrict__ ctx, _Float16* __restrict__ qf,
    _Float16* __restrict__ kf, _Float16* __restrict__ vf) {
  int idx = blockIdx.x * 256 + threadIdx.x;  // < L*D exactly
  int l = idx / DMODEL;
  int d = idx - l * DMODEL;
  int h = d >> 6, e = d & 63;
  size_t sb = (size_t)l * 2 * DMODEL;
  float sa = softplusf(sem[sb + d]);
  float sp = sem[sb + DMODEL + d];
  float ca = softplusf(ctx[sb + d]);
  float cp = ctx[sb + DMODEL + d];
  float gate = 1.0f / (1.0f + expf(-(sa * ca * cosf(sp - cp))));
  size_t qb = (size_t)l * 3 * DMODEL;
  float q = qkv[qb + d];
  float k = qkv[qb + DMODEL + d] * gate;
  float v = qkv[qb + 2 * DMODEL + d];
  size_t ho = ((size_t)h * LSEQ + l) * DHEAD + e;
  qf[ho] = (_Float16)(q > 0.0f ? q + 1.0f : expf(q));  // elu(x)+1
  kf[ho] = (_Float16)(k > 0.0f ? k + 1.0f : expf(k));
  vf[ho] = (_Float16)v;
}

// ---------------------------------------------------------------------------
// Kernel: chunked causal linear-attention scan.  One workgroup per head,
// sequential over 32 chunks of 64 tokens; all per-chunk contractions via WMMA.
//   A    = tril(Q K^T)                    (64x64, contraction DH=64)
//   out  = A @ V + Q @ S_prev             (64x64, contraction 64)
//   S   += K^T @ V                        (64x64, contraction tokens=64)
//   den  = rowsum(A) + q . kc_prev + eps
// ---------------------------------------------------------------------------
__global__ __launch_bounds__(256) void scan_kernel(
    const _Float16* __restrict__ qf, const _Float16* __restrict__ kf,
    const _Float16* __restrict__ vf, _Float16* __restrict__ attn) {
  __shared__ __align__(32) _Float16 sQ[64][64];   // [tok][dh]
  __shared__ __align__(32) _Float16 sK[64][64];   // [tok][dh]  (== B^T for QK^T)
  __shared__ __align__(32) _Float16 sVT[64][64];  // [e][tok]   (B^T for *@V)
  __shared__ __align__(32) _Float16 sAm[64][64];  // masked attn f16 [i][j]
  __shared__ __align__(32) _Float16 sST[64][64];  // S^T f16 [e][dh] (B^T for Q@S)
  __shared__ float fS[64][64];                    // S f32 [dh][e]
  __shared__ float kc[64];                        // cumulative k sum
  __shared__ float den[64];

  const int h = blockIdx.x;
  const int tid = threadIdx.x, wave = tid >> 5, lane = tid & 31;
  const _Float16* Qh = qf + (size_t)h * LSEQ * DHEAD;
  const _Float16* Kh = kf + (size_t)h * LSEQ * DHEAD;
  const _Float16* Vh = vf + (size_t)h * LSEQ * DHEAD;

  for (int i = tid; i < 64 * 64; i += 256) {
    fS[i >> 6][i & 63] = 0.0f;
    sST[i >> 6][i & 63] = (_Float16)0.0f;
  }
  if (tid < 64) kc[tid] = 0.0f;
  __syncthreads();

  const int ltok = tid >> 2;             // chunk-load row
  const int lcol = (tid & 3) * 16;       // chunk-load col base

  for (int c0 = 0; c0 < LSEQ; c0 += 64) {
    // ---- load chunk tiles --------------------------------------------------
    {
      size_t gb = (size_t)(c0 + ltok) * DHEAD + lcol;
      *(v8h*)&sQ[ltok][lcol]     = *(const v8h*)&Qh[gb];
      *(v8h*)&sQ[ltok][lcol + 8] = *(const v8h*)&Qh[gb + 8];
      *(v8h*)&sK[ltok][lcol]     = *(const v8h*)&Kh[gb];
      *(v8h*)&sK[ltok][lcol + 8] = *(const v8h*)&Kh[gb + 8];
      v8h v0 = *(const v8h*)&Vh[gb];
      v8h v1 = *(const v8h*)&Vh[gb + 8];
#pragma unroll
      for (int j = 0; j < 8; ++j) {
        sVT[lcol + j][ltok]     = v0[j];
        sVT[lcol + 8 + j][ltok] = v1[j];
      }
    }
    __syncthreads();

    // ---- A = tril(Q K^T) ---------------------------------------------------
#pragma unroll
    for (int t = 0; t < 2; ++t) {
      int tile = wave * 2 + t, it = tile >> 2, jt = tile & 3;
      v8f acc = {};
#pragma unroll
      for (int ks = 0; ks < 2; ++ks) {
        v16h a = frag_a(&sQ[0][0], 64, it * 16 + (lane & 15), ks * 32, lane);
        v16h b = frag_b(&sK[0][0], 64, jt * 16 + (lane & 15), ks * 32, lane);
        acc = wmma_f16(a, b, acc);
      }
      int n = jt * 16 + (lane & 15);
      int mb = it * 16 + 8 * (lane >> 4);
#pragma unroll
      for (int r = 0; r < 8; ++r) {
        int m = mb + r;
        sAm[m][n] = (_Float16)((m >= n) ? acc[r] : 0.0f);
      }
    }
    __syncthreads();

    // ---- den = eps + rowsum(A) + q . kc_prev -------------------------------
    if (tid < 64) {
      float s = 1e-6f;
#pragma unroll 8
      for (int j = 0; j < 64; ++j) s += (float)sAm[tid][j];
      float dq = 0.0f;
#pragma unroll 8
      for (int d = 0; d < 64; ++d) dq += (float)sQ[tid][d] * kc[d];
      den[tid] = s + dq;
    }
    __syncthreads();

    // ---- kc += colsum(K chunk) --------------------------------------------
    if (tid < 64) {
      float s = kc[tid];
#pragma unroll 8
      for (int tt = 0; tt < 64; ++tt) s += (float)sK[tt][tid];
      kc[tid] = s;
    }

    // ---- out = A@V + Q@S_prev ; attn = out / den --------------------------
#pragma unroll
    for (int t = 0; t < 2; ++t) {
      int tile = wave * 2 + t, it = tile >> 2, jt = tile & 3;
      v8f acc = {};
#pragma unroll
      for (int ks = 0; ks < 2; ++ks) {
        v16h a = frag_a(&sAm[0][0], 64, it * 16 + (lane & 15), ks * 32, lane);
        v16h b = frag_b(&sVT[0][0], 64, jt * 16 + (lane & 15), ks * 32, lane);
        acc = wmma_f16(a, b, acc);
      }
#pragma unroll
      for (int ks = 0; ks < 2; ++ks) {
        v16h a = frag_a(&sQ[0][0], 64, it * 16 + (lane & 15), ks * 32, lane);
        v16h b = frag_b(&sST[0][0], 64, jt * 16 + (lane & 15), ks * 32, lane);
        acc = wmma_f16(a, b, acc);
      }
      int n = jt * 16 + (lane & 15);
      int mb = it * 16 + 8 * (lane >> 4);
#pragma unroll
      for (int r = 0; r < 8; ++r) {
        int m = mb + r;
        attn[(size_t)(c0 + m) * DMODEL + h * DHEAD + n] =
            (_Float16)(acc[r] / den[m]);
      }
    }
    __syncthreads();

    // ---- S += K^T V --------------------------------------------------------
#pragma unroll
    for (int t = 0; t < 2; ++t) {
      int tile = wave * 2 + t, it = tile >> 2, jt = tile & 3;
      int mdh = it * 16 + (lane & 15);
      int n = jt * 16 + (lane & 15);
      int mb = it * 16 + 8 * (lane >> 4);
      v8f acc;
#pragma unroll
      for (int r = 0; r < 8; ++r) acc[r] = fS[mb + r][n];
#pragma unroll
      for (int ks = 0; ks < 2; ++ks) {
        v16h a;  // K^T fragment gathered element-wise from token-major sK
#pragma unroll
        for (int hh = 0; hh < 16; ++hh) {
          int kt = ks * 32 + (hh & 7) + 8 * (lane >> 4) + 16 * (hh >> 3);
          a[hh] = sK[kt][mdh];
        }
        v16h b = frag_b(&sVT[0][0], 64, n, ks * 32, lane);
        acc = wmma_f16(a, b, acc);
      }
#pragma unroll
      for (int r = 0; r < 8; ++r) {
        int m = mb + r;
        fS[m][n] = acc[r];
        sST[n][m] = (_Float16)acc[r];
      }
    }
    __syncthreads();
  }
}

// ---------------------------------------------------------------------------
// Host launcher.
// ---------------------------------------------------------------------------
extern "C" void kernel_launch(void* const* d_in, const int* in_sizes, int n_in,
                              void* d_out, int out_size, void* d_ws,
                              size_t ws_size, hipStream_t stream) {
  (void)in_sizes; (void)n_in; (void)out_size; (void)ws_size;
  const float* x      = (const float*)d_in[0];
  const float* W_qkv  = (const float*)d_in[1];
  const float* b_qkv  = (const float*)d_in[2];
  const float* W_sem  = (const float*)d_in[3];
  const float* b_sem  = (const float*)d_in[4];
  const float* W_ctx  = (const float*)d_in[5];
  const float* b_ctx  = (const float*)d_in[6];
  const float* W_proj = (const float*)d_in[7];
  const float* b_proj = (const float*)d_in[8];
  const float* ln_g   = (const float*)d_in[9];
  const float* ln_b   = (const float*)d_in[10];
  float* out = (float*)d_out;

  const size_t L = LSEQ, D = DMODEL;
  char* ws = (char*)d_ws;
  size_t off = 0;
  auto alloc = [&](size_t bytes) -> void* {
    void* p = ws + off;
    off += (bytes + 255) & ~(size_t)255;
    return p;
  };
  _Float16* x_h    = (_Float16*)alloc(L * D * 2);
  _Float16* xn_h   = (_Float16*)alloc(L * D * 2);
  _Float16* Wqkv_h = (_Float16*)alloc(D * 3 * D * 2);
  _Float16* Wsem_h = (_Float16*)alloc(D * 2 * D * 2);
  _Float16* Wctx_h = (_Float16*)alloc(D * 2 * D * 2);
  _Float16* Wprj_h = (_Float16*)alloc(D * D * 2);
  float*    qkv    = (float*)   alloc(L * 3 * D * 4);
  float*    sem    = (float*)   alloc(L * 2 * D * 4);
  float*    ctx    = (float*)   alloc(L * 2 * D * 4);
  _Float16* qf     = (_Float16*)alloc(L * D * 2);
  _Float16* kf     = (_Float16*)alloc(L * D * 2);
  _Float16* vf     = (_Float16*)alloc(L * D * 2);
  _Float16* attn_h = (_Float16*)alloc(L * D * 2);

  // Weight conversions
  cvt_f16_kernel<<<(int)((D * 3 * D + 255) / 256), 256, 0, stream>>>(W_qkv, Wqkv_h, (int)(D * 3 * D));
  cvt_f16_kernel<<<(int)((D * 2 * D + 255) / 256), 256, 0, stream>>>(W_sem, Wsem_h, (int)(D * 2 * D));
  cvt_f16_kernel<<<(int)((D * 2 * D + 255) / 256), 256, 0, stream>>>(W_ctx, Wctx_h, (int)(D * 2 * D));
  cvt_f16_kernel<<<(int)((D * D + 255) / 256), 256, 0, stream>>>(W_proj, Wprj_h, (int)(D * D));

  // LayerNorm + x/x_norm f16
  layernorm_cvt_kernel<<<(int)L, 256, 0, stream>>>(x, ln_g, ln_b, x_h, xn_h);

  // Projections (block tile 128x128)
  gemm_wmma_kernel<<<dim3((int)(L / 128), (int)(3 * D / 128)), 256, 0, stream>>>(
      xn_h, Wqkv_h, b_qkv, qkv, (int)L, (int)(3 * D), (int)D);
  gemm_wmma_kernel<<<dim3((int)(L / 128), (int)(2 * D / 128)), 256, 0, stream>>>(
      x_h, Wsem_h, b_sem, sem, (int)L, (int)(2 * D), (int)D);
  gemm_wmma_kernel<<<dim3((int)(L / 128), (int)(2 * D / 128)), 256, 0, stream>>>(
      x_h, Wctx_h, b_ctx, ctx, (int)L, (int)(2 * D), (int)D);

  // Gate + feature maps
  gate_feat_kernel<<<(int)(L * D / 256), 256, 0, stream>>>(qkv, sem, ctx, qf, kf, vf);

  // Chunked causal linear-attention scan (one WG per head)
  scan_kernel<<<NHEAD, 256, 0, stream>>>(qf, kf, vf, attn_h);

  // Output projection
  gemm_wmma_kernel<<<dim3((int)(L / 128), (int)(D / 128)), 256, 0, stream>>>(
      attn_h, Wprj_h, b_proj, out, (int)L, (int)D, (int)D);
}
